// TransformerXL_55851754717771
// MI455X (gfx1250) — matrix operand order repaired and checked by reading, not compiled, not run on hardware
//
#include <hip/hip_runtime.h>

#define B_    8
#define S_    2048
#define D_    1024
#define DOUT_ 1024
#define ROWS  (B_ * S_)      // 16384
#define KC    128            // key chunk in attention

typedef __attribute__((ext_vector_type(16))) __bf16 v16bf;
typedef __attribute__((ext_vector_type(8)))  float  v8f;

union Frag { v16bf v; uint4 u[2]; };

#define WMMA_BF16(a, b, c) \
  __builtin_amdgcn_wmma_f32_16x16x32_bf16(false, (a), false, (b), (short)0, (c), false, false)

__device__ __forceinline__ unsigned short f2bf(float x) {
  union { float f; unsigned u; } c; c.f = x;
  unsigned u = c.u;
  unsigned r = (u + 0x7FFFu + ((u >> 16) & 1u)) >> 16;   // round-to-nearest-even
  return (unsigned short)r;
}

// A fragment (16x32 bf16, M x K). Lane l<16: row=l, K = {k0..k0+7, k0+16..k0+23}
// Lane l>=16: row=l-16, K = {k0+8..k0+15, k0+24..k0+31}. (ISA 7.12.2)
__device__ __forceinline__ v16bf loadA(const unsigned short* base, int row0, int ld,
                                       int k0, int lane) {
  const int r    = row0 + (lane & 15);
  const int koff = k0 + ((lane >> 4) << 3);
  const unsigned short* p = base + (size_t)r * ld + koff;
  Frag f;
  f.u[0] = *reinterpret_cast<const uint4*>(p);
  f.u[1] = *reinterpret_cast<const uint4*>(p + 16);
  return f.v;
}

// B fragment (32x16 bf16, K x N), stored N-major (base + n*ld + k).
// Lane l<16: col=l, K = {k0..k0+7, k0+16..k0+23} ; lane l>=16: col=l-16, K = {k0+8..k0+15, k0+24..k0+31}.
__device__ __forceinline__ v16bf loadB(const unsigned short* base, int col0, int ld,
                                       int k0, int lane) {
  const int c    = col0 + (lane & 15);
  const int koff = k0 + ((lane >> 4) << 3);
  const unsigned short* p = base + (size_t)c * ld + koff;
  Frag f;
  f.u[0] = *reinterpret_cast<const uint4*>(p);
  f.u[1] = *reinterpret_cast<const uint4*>(p + 16);
  return f.v;
}

// ---------------- fp32 -> bf16 elementwise convert (x4 vectorized) ----------
__global__ void cvt_kernel(const float* __restrict__ src,
                           unsigned short* __restrict__ dst, int n4) {
  int i = blockIdx.x * blockDim.x + threadIdx.x;
  if (i >= n4) return;
  float4 v = reinterpret_cast<const float4*>(src)[i];
  uint2 pk;
  pk.x = (unsigned)f2bf(v.x) | ((unsigned)f2bf(v.y) << 16);
  pk.y = (unsigned)f2bf(v.z) | ((unsigned)f2bf(v.w) << 16);
  reinterpret_cast<uint2*>(dst)[i] = pk;
}

// ---------------- relative shift of pos_embed, fused with bf16 convert -----
__global__ void shift_cvt_kernel(const float* __restrict__ pos,
                                 unsigned short* __restrict__ dst) {
  int i = blockIdx.x * blockDim.x + threadIdx.x;   // over ROWS*D/4
  int idx4  = i << 2;
  int d4    = idx4 & (D_ - 1);
  int rowid = idx4 >> 10;                          // = b*S + s
  int ii = rowid >> 3;                             // / B
  int jj = rowid & 7;                              // % B
  int f  = (ii + 1) * B_ + jj;                     // row in padded (B,S+1) flat
  int bp = f / (S_ + 1);
  int sp = f - bp * (S_ + 1);
  uint2 pk;
  if (sp == 0) {
    pk.x = 0u; pk.y = 0u;
  } else {
    const float4 v = *reinterpret_cast<const float4*>(
        pos + (((size_t)bp * S_ + (sp - 1)) << 10) + d4);
    pk.x = (unsigned)f2bf(v.x) | ((unsigned)f2bf(v.y) << 16);
    pk.y = (unsigned)f2bf(v.z) | ((unsigned)f2bf(v.w) << 16);
  }
  reinterpret_cast<uint2*>(dst)[i] = pk;
}

// ---------------- generic bf16 WMMA GEMM: C(M x N) = A(M x K) * B(N x K)^T --
// Workgroup tile 64(M) x 256(N); 8 waves in 2x4; each wave: 2x4 16x16 tiles.
// mode 0: qkv epilogue (q+bq, k, k+bk, v transposed) ; mode 1: bf16 + bias ;
// mode 2: fp32 out.
__global__ __launch_bounds__(256) void gemm_kernel(
    const unsigned short* __restrict__ A, const unsigned short* __restrict__ Bm,
    int K, int mode,
    const float* __restrict__ bq, const float* __restrict__ bk,
    unsigned short* __restrict__ o_qb, unsigned short* __restrict__ o_k,
    unsigned short* __restrict__ o_kb, unsigned short* __restrict__ o_vT,
    unsigned short* __restrict__ o_bf, float* __restrict__ o_f) {
  const int lane = threadIdx.x & 31;
  const int w    = threadIdx.x >> 5;
  const int m0   = blockIdx.y * 64  + (w >> 2) * 32;
  const int n0   = blockIdx.x * 256 + (w & 3) * 64;

  const v8f z = {0.f, 0.f, 0.f, 0.f, 0.f, 0.f, 0.f, 0.f};
  v8f acc[2][4];
#pragma unroll
  for (int i = 0; i < 2; i++)
#pragma unroll
    for (int j = 0; j < 4; j++) acc[i][j] = z;

  for (int k0 = 0; k0 < K; k0 += 32) {
    v16bf a0 = loadA(A, m0,      K, k0, lane);
    v16bf a1 = loadA(A, m0 + 16, K, k0, lane);
#pragma unroll
    for (int j = 0; j < 4; j++) {
      v16bf b = loadB(Bm, n0 + 16 * j, K, k0, lane);
      acc[0][j] = WMMA_BF16(a0, b, acc[0][j]);
      acc[1][j] = WMMA_BF16(a1, b, acc[1][j]);
    }
  }

  const int hi = lane >> 4, nn = lane & 15;
#pragma unroll
  for (int i = 0; i < 2; i++)
#pragma unroll
    for (int j = 0; j < 4; j++) {
      const int e = n0 + 16 * j + nn;
#pragma unroll
      for (int r = 0; r < 8; r++) {
        const int row = m0 + 16 * i + r + hi * 8;
        const float v = acc[i][j][r];
        if (mode == 0) {
          if (e < 1024) {
            o_qb[(size_t)row * 1024 + e] = f2bf(v + bq[e]);
          } else if (e < 2048) {
            const int ec = e - 1024;
            o_k [(size_t)row * 1024 + ec] = f2bf(v);
            o_kb[(size_t)row * 1024 + ec] = f2bf(v + bk[ec]);
          } else {
            o_vT[(size_t)(e - 2048) * ROWS + row] = f2bf(v);   // V transposed
          }
        } else if (mode == 1) {
          o_bf[(size_t)row * 1024 + e] = f2bf(v + bq[e]);      // bq = r_net_b
        } else {
          o_f[(size_t)row * 1024 + e] = v;
        }
      }
    }
}

// ---------------- fused relative attention (flash style) -------------------
// One workgroup = 16 query rows; 8 waves. score[q,k] = qb[q]·k[k] + kb[q]·pk[k]
__global__ __launch_bounds__(256) void attn_kernel(
    const unsigned short* __restrict__ qb, const unsigned short* __restrict__ kmat,
    const unsigned short* __restrict__ kb, const unsigned short* __restrict__ pkey,
    const unsigned short* __restrict__ vT, unsigned short* __restrict__ ctx) {
  __shared__ uint4 sh_q4 [16 * 1024 / 8];   // 32 KB: q + bias_q rows (bf16)
  __shared__ uint4 sh_kb4[16 * 1024 / 8];   // 32 KB: k + bias_k rows (bf16)
  __shared__ uint4 sh_p4 [16 * KC / 8];     //  4 KB: P chunk (bf16)
  __shared__ float sh_s[16 * KC];           //  8 KB: raw scores chunk
  __shared__ float sh_m[16], sh_l[16], sh_f[16];

  const int tid  = threadIdx.x;
  const int lane = tid & 31, w = tid >> 5;
  const int b    = blockIdx.x >> 7;                 // / (S/16)
  const int qt   = blockIdx.x & 127;
  const int qrow0 = b * S_ + qt * 16;

  {  // stage query-side rows once: async global->LDS on CDNA5, fallback copy
    const uint4* gq = reinterpret_cast<const uint4*>(qb + (size_t)qrow0 * 1024);
    const uint4* gk = reinterpret_cast<const uint4*>(kb + (size_t)qrow0 * 1024);
#if defined(__gfx1250__) && __has_builtin(__builtin_amdgcn_global_load_async_to_lds_b128)
    typedef int gvec4 __attribute__((vector_size(16)));
    typedef __attribute__((address_space(1))) gvec4* as1v4;
    typedef __attribute__((address_space(3))) gvec4* as3v4;
    for (int i = tid; i < 16 * 1024 / 8; i += 256) {
      __builtin_amdgcn_global_load_async_to_lds_b128(
          (as1v4)(gq + i), (as3v4)(sh_q4 + i), 0, 0);
      __builtin_amdgcn_global_load_async_to_lds_b128(
          (as1v4)(gk + i), (as3v4)(sh_kb4 + i), 0, 0);
    }
#if __has_builtin(__builtin_amdgcn_s_wait_asynccnt)
    __builtin_amdgcn_s_wait_asynccnt(0);
#else
    asm volatile("s_wait_asynccnt 0" ::: "memory");
#endif
#else
    for (int i = tid; i < 16 * 1024 / 8; i += 256) { sh_q4[i] = gq[i]; sh_kb4[i] = gk[i]; }
#endif
  }
  if (tid < 16) { sh_m[tid] = -3.0e38f; sh_l[tid] = 0.f; sh_f[tid] = 1.f; }
  __syncthreads();

  const unsigned short* shq  = reinterpret_cast<const unsigned short*>(sh_q4);
  const unsigned short* shkb = reinterpret_cast<const unsigned short*>(sh_kb4);
  const unsigned short* shp  = reinterpret_cast<const unsigned short*>(sh_p4);

  const v8f z = {0.f, 0.f, 0.f, 0.f, 0.f, 0.f, 0.f, 0.f};
  v8f accO[8];
#pragma unroll
  for (int nt = 0; nt < 8; nt++) accO[nt] = z;

  const unsigned short* kbase = kmat + (size_t)b * S_ * 1024;
  const unsigned short* pbase = pkey + (size_t)b * S_ * 1024;
  const int hi = lane >> 4, nn = lane & 15;

  for (int kc0 = 0; kc0 < S_; kc0 += KC) {
    const int keycol0 = kc0 + 16 * w;

    // prefetch next chunk's key-side panels into GL2/L0 while we compute
    if (kc0 + KC < S_) {
      const int pcol = keycol0 + KC + (lane & 15);
      __builtin_prefetch(kbase + (size_t)pcol * 1024 + ((lane >> 4) << 9), 0, 1);
      __builtin_prefetch(pbase + (size_t)pcol * 1024 + ((lane >> 4) << 9), 0, 1);
    }

    // ---- Phase A: 16x16 score tile per wave, AC + BD fused into one acc ----
    v8f sc = z;
#pragma unroll 8
    for (int kk = 0; kk < 32; kk++) {
      const int k0 = kk * 32;
      v16bf aq = loadA(shq, 0, 1024, k0, lane);
      v16bf bk_ = loadB(kbase, keycol0, 1024, k0, lane);
      sc = WMMA_BF16(aq, bk_, sc);
      v16bf ak = loadA(shkb, 0, 1024, k0, lane);
      v16bf bp = loadB(pbase, keycol0, 1024, k0, lane);
      sc = WMMA_BF16(ak, bp, sc);
    }
#pragma unroll
    for (int r = 0; r < 8; r++)
      sh_s[(r + hi * 8) * KC + 16 * w + nn] = sc[r] * 0.03125f;  // 1/sqrt(1024)
    __syncthreads();

    // ---- Phase B: online softmax over the 128-key chunk --------------------
    {
      const int row = tid >> 4, j = tid & 15;
      float4 s0 = *reinterpret_cast<const float4*>(&sh_s[row * KC + j * 8]);
      float4 s1 = *reinterpret_cast<const float4*>(&sh_s[row * KC + j * 8 + 4]);
      float mx = fmaxf(fmaxf(fmaxf(s0.x, s0.y), fmaxf(s0.z, s0.w)),
                       fmaxf(fmaxf(s1.x, s1.y), fmaxf(s1.z, s1.w)));
#pragma unroll
      for (int off = 8; off >= 1; off >>= 1) mx = fmaxf(mx, __shfl_xor(mx, off, 16));
      const float m_old = sh_m[row];
      const float m_new = fmaxf(m_old, mx);
      float p0 = __expf(s0.x - m_new), p1 = __expf(s0.y - m_new);
      float p2 = __expf(s0.z - m_new), p3 = __expf(s0.w - m_new);
      float p4 = __expf(s1.x - m_new), p5 = __expf(s1.y - m_new);
      float p6 = __expf(s1.z - m_new), p7 = __expf(s1.w - m_new);
      float sum = ((p0 + p1) + (p2 + p3)) + ((p4 + p5) + (p6 + p7));
#pragma unroll
      for (int off = 8; off >= 1; off >>= 1) sum += __shfl_xor(sum, off, 16);
      uint4 pw;
      pw.x = (unsigned)f2bf(p0) | ((unsigned)f2bf(p1) << 16);
      pw.y = (unsigned)f2bf(p2) | ((unsigned)f2bf(p3) << 16);
      pw.z = (unsigned)f2bf(p4) | ((unsigned)f2bf(p5) << 16);
      pw.w = (unsigned)f2bf(p6) | ((unsigned)f2bf(p7) << 16);
      sh_p4[(row * KC + j * 8) >> 3] = pw;
      if (j == 0) {
        const float fct = __expf(m_old - m_new);
        sh_f[row] = fct;
        sh_l[row] = sh_l[row] * fct + sum;
        sh_m[row] = m_new;
      }
    }
    __syncthreads();

    // ---- Phase C: rescale running ctx, then P @ V (wave owns 128 cols) ----
    float srow[8];
#pragma unroll
    for (int r = 0; r < 8; r++) srow[r] = sh_f[r + hi * 8];
#pragma unroll
    for (int nt = 0; nt < 8; nt++)
#pragma unroll
      for (int r = 0; r < 8; r++) accO[nt][r] *= srow[r];

    v16bf pf[4];
#pragma unroll
    for (int kk = 0; kk < 4; kk++) pf[kk] = loadA(shp, 0, KC, kk * 32, lane);
    const unsigned short* vb = vT + (size_t)b * S_ + kc0;   // vT[col*ROWS + key]
#pragma unroll
    for (int nt = 0; nt < 8; nt++) {
      const int col0 = 128 * w + 16 * nt;
#pragma unroll
      for (int kk = 0; kk < 4; kk++) {
        v16bf bv = loadB(vb, col0, ROWS, kk * 32, lane);
        accO[nt] = WMMA_BF16(pf[kk], bv, accO[nt]);
      }
    }
    __syncthreads();
  }

  // ---- finalize: divide by softmax denominator, emit bf16 ctx -------------
  float linv[8];
#pragma unroll
  for (int r = 0; r < 8; r++) linv[r] = 1.f / sh_l[r + hi * 8];
#pragma unroll
  for (int nt = 0; nt < 8; nt++) {
    const int col = 128 * w + 16 * nt + nn;
#pragma unroll
    for (int r = 0; r < 8; r++) {
      const int row = qrow0 + r + hi * 8;
      ctx[(size_t)row * 1024 + col] = f2bf(accO[nt][r] * linv[r]);
    }
  }
}

// ---------------------------------------------------------------------------
extern "C" void kernel_launch(void* const* d_in, const int* in_sizes, int n_in,
                              void* d_out, int out_size, void* d_ws, size_t ws_size,
                              hipStream_t stream) {
  const float* word   = (const float*)d_in[0];
  const float* pos    = (const float*)d_in[1];
  const float* bias_q = (const float*)d_in[2];
  const float* bias_k = (const float*)d_in[3];
  const float* W_qkv  = (const float*)d_in[4];
  const float* rnet_w = (const float*)d_in[5];
  const float* rnet_b = (const float*)d_in[6];
  const float* W_out  = (const float*)d_in[7];
  float* out = (float*)d_out;

  unsigned short* ws = (unsigned short*)d_ws;
  size_t off = 0;
  auto alloc = [&](size_t n) { unsigned short* p = ws + off; off += n; return p; };
  const size_t ND = (size_t)ROWS * D_;
  unsigned short* we   = alloc(ND);               // word_embed bf16
  unsigned short* shft = alloc(ND);               // shifted pos bf16
  unsigned short* wqkv = alloc((size_t)3 * DOUT_ * D_);
  unsigned short* rnet = alloc((size_t)DOUT_ * D_);
  unsigned short* wout = alloc((size_t)D_ * DOUT_);
  unsigned short* qbm  = alloc(ND);               // q + bias_q
  unsigned short* kmat = alloc(ND);               // k
  unsigned short* kbm  = alloc(ND);               // k + bias_k
  unsigned short* vTm  = alloc(ND);               // v, transposed [e][row]
  unsigned short* pkm  = alloc(ND);               // pos_key
  unsigned short* ctxm = alloc(ND);               // attention context
  (void)in_sizes; (void)n_in; (void)out_size; (void)ws_size;

  // 1. conversions + relative shift
  cvt_kernel<<<(ND / 4) / 256, 256, 0, stream>>>(word, we, (int)(ND / 4));
  cvt_kernel<<<(3 * DOUT_ * D_ / 4) / 256, 256, 0, stream>>>(W_qkv, wqkv, 3 * DOUT_ * D_ / 4);
  cvt_kernel<<<(DOUT_ * D_ / 4) / 256, 256, 0, stream>>>(rnet_w, rnet, DOUT_ * D_ / 4);
  cvt_kernel<<<(D_ * DOUT_ / 4) / 256, 256, 0, stream>>>(W_out, wout, D_ * DOUT_ / 4);
  shift_cvt_kernel<<<(ND / 4) / 256, 256, 0, stream>>>(pos, shft);

  // 2. qkv projection with fused bias / transpose-V epilogue
  dim3 gq(3 * DOUT_ / 256, ROWS / 64);
  gemm_kernel<<<gq, 256, 0, stream>>>(we, wqkv, D_, 0, bias_q, bias_k,
                                      qbm, kmat, kbm, vTm, nullptr, nullptr);

  // 3. pos_key = shifted @ r_net_w^T + r_net_b
  dim3 gp(DOUT_ / 256, ROWS / 64);
  gemm_kernel<<<gp, 256, 0, stream>>>(shft, rnet, D_, 1, rnet_b, nullptr,
                                      nullptr, nullptr, nullptr, nullptr, pkm, nullptr);

  // 4. fused relative attention
  attn_kernel<<<ROWS / 16, 256, 0, stream>>>(qbm, kmat, kbm, pkm, vTm, ctxm);

  // 5. out = ctx @ W_out^T (fp32 result)
  dim3 go(D_ / 256, ROWS / 64);
  gemm_kernel<<<go, 256, 0, stream>>>(ctxm, wout, DOUT_, 2, nullptr, nullptr,
                                      nullptr, nullptr, nullptr, nullptr, nullptr, out);
}
